// _GCNEncoder_43258910605759
// MI455X (gfx1250) — matrix operand order repaired and checked by reading, not compiled, not run on hardware
//
#include <hip/hip_runtime.h>
#include <hip/hip_bf16.h>
#include <stdint.h>

// CDNA5 / gfx1250: wave32, v_wmma_f32_16x16x32_bf16 (f32 accumulate).
typedef __attribute__((ext_vector_type(16))) __bf16 v16bf;
typedef __attribute__((ext_vector_type(8)))  __bf16 v8bf;
typedef __attribute__((ext_vector_type(8)))  float  v8f;

#define C 128           // feature width (both GCN layers)
#define WAVE 32

// ---------------------------------------------------------------------------
// deg[n] = 1 (self loop) + sum_{e: dst==n} ew[e];  dinv = rsqrt(deg)
// ---------------------------------------------------------------------------
__global__ void k_init_deg(float* __restrict__ deg, int N) {
  int n = blockIdx.x * blockDim.x + threadIdx.x;
  if (n < N) deg[n] = 1.0f;                        // self-loop weight
}

__global__ void k_accum_deg(float* __restrict__ deg,
                            const long long* __restrict__ ei,   // [2,E] int64
                            const float* __restrict__ ew, int E) {
  int e = blockIdx.x * blockDim.x + threadIdx.x;
  if (e < E) {
    long long d = ei[(long long)E + e];
    atomicAdd(&deg[d], ew[e]);
  }
}

__global__ void k_dinv(const float* __restrict__ deg, float* __restrict__ dinv, int N) {
  int n = blockIdx.x * blockDim.x + threadIdx.x;
  if (n < N) {
    float dg = deg[n];
    dinv[n] = dg > 0.0f ? rsqrtf(dg) : 0.0f;
  }
}

// ---------------------------------------------------------------------------
// Elementwise fp32 -> bf16 hi/lo split:  f ~= hi + lo, |err| ~ 2^-16 |f|
// ---------------------------------------------------------------------------
__global__ void k_split(const float* __restrict__ src,
                        __bf16* __restrict__ hi, __bf16* __restrict__ lo, int n) {
  int i = blockIdx.x * blockDim.x + threadIdx.x;
  if (i < n) {
    float f = src[i];
    __bf16 h = (__bf16)f;
    hi[i] = h;
    lo[i] = (__bf16)(f - (float)h);
  }
}

// Split + transpose the 128x128 weight: Wt[n][k] = W[k][n], so a B-operand
// lane (one output column, 16 consecutive K) reads a contiguous 32B run.
__global__ void k_split_w_t(const float* __restrict__ W,
                            __bf16* __restrict__ wth, __bf16* __restrict__ wtl) {
  int i = blockIdx.x * blockDim.x + threadIdx.x;   // i over C*C
  if (i < C * C) {
    int k = i >> 7, n = i & 127;
    float f = W[i];                                 // W[k][n]
    __bf16 h = (__bf16)f;
    wth[n * C + k] = h;
    wtl[n * C + k] = (__bf16)(f - (float)h);
  }
}

// ---------------------------------------------------------------------------
// H = X * W using pre-split bf16 operands:
//   acc += Xh*Wh + Xh*Wl + Xl*Wh   (3-term fp32 emulation, f32 accumulate)
// One wave per 16x16 D tile; 8 waves/block cover the 8 column tiles; K=128
// in 4 blocks of 32 -> 12 v_wmma per wave, 4 packed loads per K-block.
// ---------------------------------------------------------------------------
__global__ void k_gemm_wmma(const __bf16* __restrict__ Xh,
                            const __bf16* __restrict__ Xl,
                            const __bf16* __restrict__ Wth,
                            const __bf16* __restrict__ Wtl,
                            float* __restrict__ H) {
  const int wave = threadIdx.x >> 5;      // 0..7 -> N tile
  const int lane = threadIdx.x & 31;
  const int mt   = blockIdx.x;            // M tile
  const int l16  = lane & 15;
  const int hi   = lane >> 4;             // half-wave select

  const int arow = mt * 16 + l16;         // A row held by this lane
  const int bcol = wave * 16 + l16;       // B/D column held by this lane

  v8f acc = {};

#pragma unroll
  for (int k0 = 0; k0 < C; k0 += 32) {
    // A: lanes 0-15 slots = K{k0..k0+15}, lanes 16-31 slots = K{k0+16..k0+31}
    const int abase = arow * C + k0 + hi * 16;
    v16bf ah = *(const v16bf*)(Xh + abase);
    v16bf al = *(const v16bf*)(Xl + abase);

    // B: lane holds one column; lanes 0-15 carry K[k0..k0+15], lanes 16-31
    // carry K[k0+16..k0+31]; contiguous in the transposed weight.
    const int bbase = bcol * C + k0 + hi * 16;
    v16bf bh = *(const v16bf*)(Wth + bbase);
    v16bf bl = *(const v16bf*)(Wtl + bbase);

    acc = __builtin_amdgcn_wmma_f32_16x16x32_bf16(false, ah, false, bh,
                                                  (short)0, acc, false, false);
    acc = __builtin_amdgcn_wmma_f32_16x16x32_bf16(false, ah, false, bl,
                                                  (short)0, acc, false, false);
    acc = __builtin_amdgcn_wmma_f32_16x16x32_bf16(false, al, false, bh,
                                                  (short)0, acc, false, false);
  }

  // D layout: VGPR i -> row mt*16 + i + 8*hi, col bcol
  float* hp = H + (long long)(mt * 16) * C + bcol;
#pragma unroll
  for (int i = 0; i < 8; ++i)
    hp[(long long)(i + hi * 8) * C] = acc[i];
}

// ---------------------------------------------------------------------------
// out[n,c] = b[c] + H[n,c] * dinv[n]^2   (bias + self-loop), float4 streams
// ---------------------------------------------------------------------------
__global__ void k_init_out(const float4* __restrict__ H4,
                           const float* __restrict__ dinv,
                           const float4* __restrict__ b4,
                           float4* __restrict__ out4, int NC4) {
  int i = blockIdx.x * blockDim.x + threadIdx.x;
  if (i < NC4) {
    int n  = i >> 5;                      // 32 float4 per 128-wide row
    int c4 = i & 31;
    float dv = dinv[n];
    float s = dv * dv;
    float4 h = H4[i], b = b4[c4];
    out4[i] = make_float4(b.x + h.x * s, b.y + h.y * s,
                          b.z + h.z * s, b.w + h.w * s);
  }
}

// ---------------------------------------------------------------------------
// Edge scatter: out[dst] += H[src] * (dinv[src]*ew*dinv[dst])
// One wave per edge; lane l covers features {l, l+32, l+64, l+96} so every
// load/atomic instruction touches one contiguous 128B region (L2-friendly).
// ---------------------------------------------------------------------------
__global__ void k_scatter(const float* __restrict__ H,
                          const float* __restrict__ dinv,
                          const long long* __restrict__ ei,
                          const float* __restrict__ ew,
                          float* __restrict__ out, int E) {
  int t = blockIdx.x * blockDim.x + threadIdx.x;
  int e = t >> 5;
  if (e >= E) return;
  int c = t & 31;

  long long s = ei[e];
  long long d = ei[(long long)E + e];
  float nrm = dinv[s] * ew[e] * dinv[d];

  const float* hp = H + s * C + c;
  float* op = out + d * C + c;
#pragma unroll
  for (int j = 0; j < 4; ++j)
    atomicAdd(op + 32 * j, hp[32 * j] * nrm);
}

__global__ void k_relu(float4* __restrict__ out4, int NC4) {
  int i = blockIdx.x * blockDim.x + threadIdx.x;
  if (i < NC4) {
    float4 v = out4[i];
    out4[i] = make_float4(fmaxf(v.x, 0.f), fmaxf(v.y, 0.f),
                          fmaxf(v.z, 0.f), fmaxf(v.w, 0.f));
  }
}

// ---------------------------------------------------------------------------
extern "C" void kernel_launch(void* const* d_in, const int* in_sizes, int n_in,
                              void* d_out, int out_size, void* d_ws, size_t ws_size,
                              hipStream_t stream) {
  const float*     x  = (const float*)d_in[0];
  const long long* ei = (const long long*)d_in[1];   // int64 [2,E]
  const float*     ew = (const float*)d_in[2];
  const float*     W1 = (const float*)d_in[3];
  const float*     b1 = (const float*)d_in[4];
  const float*     W2 = (const float*)d_in[5];
  const float*     b2 = (const float*)d_in[6];

  const int N   = in_sizes[0] / C;     // 50000 (divisible by 16)
  const int E   = in_sizes[2];         // 600000
  const int NC  = N * C;
  const int NC4 = NC / 4;
  const int CC  = C * C;

  float* out = (float*)d_out;

  // workspace: deg|dinv|h|a1 (f32) then xh|xl|wth|wtl (bf16)   ~= 78 MB
  float*  deg  = (float*)d_ws;
  float*  dinv = deg + N;
  float*  h    = dinv + N;
  float*  a1   = h + (size_t)NC;
  __bf16* xh   = (__bf16*)(a1 + (size_t)NC);
  __bf16* xl   = xh + (size_t)NC;
  __bf16* wth  = xl + (size_t)NC;
  __bf16* wtl  = wth + CC;

  const int TB = 256;
  const int gN   = (N   + TB - 1) / TB;
  const int gE   = (E   + TB - 1) / TB;
  const int gNC  = (NC  + TB - 1) / TB;
  const int gNC4 = (NC4 + TB - 1) / TB;
  const int gCC  = (CC  + TB - 1) / TB;
  const int gEsc = ((E * WAVE) + TB - 1) / TB;
  const int gMt  = N / 16;

  // Normalization (graph is static; shared by both layers)
  k_init_deg <<<gN,  TB, 0, stream>>>(deg, N);
  k_accum_deg<<<gE,  TB, 0, stream>>>(deg, ei, ew, E);
  k_dinv     <<<gN,  TB, 0, stream>>>(deg, dinv, N);

  // Layer 1
  k_split    <<<gNC, TB, 0, stream>>>(x, xh, xl, NC);
  k_split_w_t<<<gCC, TB, 0, stream>>>(W1, wth, wtl);
  k_gemm_wmma<<<gMt, TB, 0, stream>>>(xh, xl, wth, wtl, h);
  k_init_out <<<gNC4,TB, 0, stream>>>((const float4*)h, dinv, (const float4*)b1,
                                      (float4*)a1, NC4);
  k_scatter  <<<gEsc,TB, 0, stream>>>(h, dinv, ei, ew, a1, E);
  k_relu     <<<gNC4,TB, 0, stream>>>((float4*)a1, NC4);

  // Layer 2
  k_split    <<<gNC, TB, 0, stream>>>(a1, xh, xl, NC);
  k_split_w_t<<<gCC, TB, 0, stream>>>(W2, wth, wtl);
  k_gemm_wmma<<<gMt, TB, 0, stream>>>(xh, xl, wth, wtl, h);
  k_init_out <<<gNC4,TB, 0, stream>>>((const float4*)h, dinv, (const float4*)b2,
                                      (float4*)out, NC4);
  k_scatter  <<<gEsc,TB, 0, stream>>>(h, dinv, ei, ew, out, E);
  k_relu     <<<gNC4,TB, 0, stream>>>((float4*)out, NC4);
}
